// EncoderBlock_13469017440885
// MI455X (gfx1250) — compile-verified, matrix operand order checked by reading, not run on hardware
//
#include <hip/hip_runtime.h>
#include <stddef.h>
#include <stdint.h>

typedef unsigned short u16;
typedef __attribute__((ext_vector_type(16))) __bf16 bf16x16;
typedef __attribute__((ext_vector_type(8)))  float  v8f;
typedef __attribute__((ext_vector_type(4)))  unsigned int uint4v;

union FragU { bf16x16 v; uint4v q[2]; };

__device__ __forceinline__ u16 f2bf(float f) {
    unsigned x = __float_as_uint(f);
    unsigned r = x + 0x7FFFu + ((x >> 16) & 1u);
    return (u16)(r >> 16);
}

__device__ __forceinline__ v8f wmma_bf16(FragU a, FragU b, v8f c) {
    return __builtin_amdgcn_wmma_f32_16x16x32_bf16(
        false, a.v, false, b.v, (short)0, c, false, false);
}

// DPP cross-lane combine within each 16-lane half (no LDS traffic).
#define DPPF(x, ctrl) \
    __int_as_float(__builtin_amdgcn_mov_dpp(__float_as_int(x), (ctrl), 0xf, 0xf, true))

__device__ __forceinline__ float red16_max(float v) {
    v = fmaxf(v, DPPF(v, 0xB1));   // quad_perm(1,0,3,2)  : xor 1
    v = fmaxf(v, DPPF(v, 0x4E));   // quad_perm(2,3,0,1)  : xor 2
    v = fmaxf(v, DPPF(v, 0x141));  // row_half_mirror     : combine quads
    v = fmaxf(v, DPPF(v, 0x140));  // row_mirror          : combine halves of 16
    return v;
}
__device__ __forceinline__ float red16_sum(float v) {
    v += DPPF(v, 0xB1);
    v += DPPF(v, 0x4E);
    v += DPPF(v, 0x141);
    v += DPPF(v, 0x140);
    return v;
}

// ---------------------------------------------------------------------------
// CDNA5 async global->LDS staging (ASYNCcnt-tracked, GV addressing mode)
// ---------------------------------------------------------------------------
__device__ __forceinline__ void async_b128(unsigned ldsOff, const void* g) {
    asm volatile("global_load_async_to_lds_b128 %0, %1, off"
                 :: "v"(ldsOff), "v"((unsigned long long)(size_t)g)
                 : "memory");
}
__device__ __forceinline__ void wait_async0() {
    asm volatile("s_wait_asynccnt 0x0" ::: "memory");
}
__device__ __forceinline__ unsigned lds_off(const void* p) {
    return (unsigned)(size_t)p;   // LDS address is the low 32 bits of the pointer
}

// ---------------------------------------------------------------------------
// Fused fp32->bf16 convert + transpose: in [K][N] f32 -> out [N][K] bf16.
// ---------------------------------------------------------------------------
__global__ __launch_bounds__(256, 2) void cvt_transpose_bf16(
    const float* __restrict__ in, u16* __restrict__ out, int K, int N)
{
    __shared__ u16 tile[32][33];
    const int k0 = blockIdx.y * 32, n0 = blockIdx.x * 32;
    const int tx = threadIdx.x & 31, ty = threadIdx.x >> 5;   // 32 x 8
#pragma unroll
    for (int i = 0; i < 32; i += 8)
        tile[ty + i][tx] = f2bf(in[(size_t)(k0 + ty + i) * N + n0 + tx]);
    __syncthreads();
#pragma unroll
    for (int i = 0; i < 32; i += 8)
        out[(size_t)(n0 + ty + i) * K + k0 + tx] = tile[tx][ty + i];
}

// ---------------------------------------------------------------------------
// LayerNorm: one 1024-wide row per 256-thread block, bf16 output.
// Faithful to reference: ddof=1 variance, eps added to sqrt(var).
// ---------------------------------------------------------------------------
__global__ __launch_bounds__(256, 2) void layernorm_bf16(
    const float* __restrict__ X, const float* __restrict__ g,
    const float* __restrict__ be, u16* __restrict__ out)
{
    const int row = blockIdx.x;
    const int tid = threadIdx.x;
    const float* xr = X + (size_t)row * 1024;
    float v[4];
    float s = 0.f, sq = 0.f;
#pragma unroll
    for (int i = 0; i < 4; ++i) {
        v[i] = xr[i * 256 + tid];
        s += v[i];
        sq += v[i] * v[i];
    }
#pragma unroll
    for (int off = 16; off > 0; off >>= 1) {
        s  += __shfl_down(s, off);
        sq += __shfl_down(sq, off);
    }
    __shared__ float rs[8], rq[8], stats[2];
    const int w = tid >> 5, lane = tid & 31;
    if (lane == 0) { rs[w] = s; rq[w] = sq; }
    __syncthreads();
    if (tid == 0) {
        float S = 0.f, Q = 0.f;
#pragma unroll
        for (int i = 0; i < 8; ++i) { S += rs[i]; Q += rq[i]; }
        float mean = S * (1.0f / 1024.0f);
        float var  = fmaxf((Q - 1024.0f * mean * mean) * (1.0f / 1023.0f), 0.f);
        stats[0] = mean;
        stats[1] = 1.0f / (sqrtf(var) + 1e-6f);
    }
    __syncthreads();
    const float mean = stats[0], inv = stats[1];
    u16* orow = out + (size_t)row * 1024;
#pragma unroll
    for (int i = 0; i < 4; ++i) {
        int c = i * 256 + tid;
        orow[c] = f2bf(g[c] * (v[i] - mean) * inv + be[c]);
    }
}

// ---------------------------------------------------------------------------
// bf16 WMMA GEMM: C[M,N] = A[M,K] @ Bt[N,K]^T + bias (+relu) (+f32 residual)
// 128x128 block tile, K step 32, 8 waves = 2(M) x 4(N), wave tile 64x32.
// Double-buffered async-to-LDS staging: next tile copies overlap WMMAs.
// ---------------------------------------------------------------------------
template<int RELU, int RESID, int OUTF32>
__global__ __launch_bounds__(256, 1) void gemm_bf16_wmma(
    const u16* __restrict__ A, const u16* __restrict__ Bt,
    const float* __restrict__ bias, const float* __restrict__ resid,
    float* __restrict__ outF, u16* __restrict__ outH,
    int M, int N, int K)
{
    __shared__ u16 As[2][128 * 40];   // [m][k], pitch 40 u16 = 80B (16B aligned)
    __shared__ u16 Bs[2][128 * 40];   // [n][k], same pitch

    const int tid  = threadIdx.x;
    const int lane = tid & 31;
    const int w    = tid >> 5;
    const int l    = lane & 15;
    const int hi   = lane >> 4;
    const int wm   = (w & 1) * 64;
    const int wn   = (w >> 1) * 32;
    const int bm0  = blockIdx.y * 128;
    const int bn0  = blockIdx.x * 128;

    v8f c[4][2];
#pragma unroll
    for (int i = 0; i < 4; ++i)
#pragma unroll
        for (int j = 0; j < 2; ++j) c[i][j] = {};

    const int srow  = tid >> 1;             // 0..127
    const int shalf = (tid & 1) * 16;       // 0 or 16 (u16 units)
    const unsigned ldsA0 = lds_off(&As[0][srow * 40 + shalf]);
    const unsigned ldsA1 = lds_off(&As[1][srow * 40 + shalf]);
    const unsigned ldsB0 = lds_off(&Bs[0][srow * 40 + shalf]);
    const unsigned ldsB1 = lds_off(&Bs[1][srow * 40 + shalf]);
    const u16* agBase = A  + (size_t)(bm0 + srow) * K + shalf;
    const u16* bgBase = Bt + (size_t)(bn0 + srow) * K + shalf;

    // prologue: stage tile 0 into buffer 0
    async_b128(ldsA0,      agBase);
    async_b128(ldsA0 + 16, agBase + 8);
    async_b128(ldsB0,      bgBase);
    async_b128(ldsB0 + 16, bgBase + 8);
    wait_async0();
    __syncthreads();

    int cur = 0;
    for (int k0 = 0; k0 < K; k0 += 32) {
        // issue next tile into the other buffer (overlaps with WMMAs below)
        if (k0 + 32 < K) {
            const unsigned la = cur ? ldsA0 : ldsA1;
            const unsigned lb = cur ? ldsB0 : ldsB1;
            async_b128(la,      agBase + k0 + 32);
            async_b128(la + 16, agBase + k0 + 40);
            async_b128(lb,      bgBase + k0 + 32);
            async_b128(lb + 16, bgBase + k0 + 40);
        }

        const u16* Ac = As[cur];
        const u16* Bc = Bs[cur];
        FragU bfr[2];
#pragma unroll
        for (int ns = 0; ns < 2; ++ns) {
            int col = wn + ns * 16 + l;
            bfr[ns].q[0] = *(const uint4v*)&Bc[col * 40 + hi * 8];
            bfr[ns].q[1] = *(const uint4v*)&Bc[col * 40 + 16 + hi * 8];
        }
#pragma unroll
        for (int sub = 0; sub < 4; ++sub) {
            FragU afr;
            int row = wm + sub * 16 + l;
            afr.q[0] = *(const uint4v*)&Ac[row * 40 + hi * 8];
            afr.q[1] = *(const uint4v*)&Ac[row * 40 + 16 + hi * 8];
#pragma unroll
            for (int ns = 0; ns < 2; ++ns)
                c[sub][ns] = wmma_bf16(afr, bfr[ns], c[sub][ns]);
        }

        wait_async0();
        __syncthreads();
        cur ^= 1;
    }

#pragma unroll
    for (int sub = 0; sub < 4; ++sub) {
#pragma unroll
        for (int ns = 0; ns < 2; ++ns) {
            int gn = bn0 + wn + ns * 16 + l;
            float bv = bias[gn];
#pragma unroll
            for (int r = 0; r < 8; ++r) {
                int gm = bm0 + wm + sub * 16 + r + hi * 8;
                float val = c[sub][ns][r] + bv;
                if (RELU) val = fmaxf(val, 0.f);
                size_t oi = (size_t)gm * N + gn;
                if (RESID) val += resid[oi];
                if (OUTF32) outF[oi] = val;
                else        outH[oi] = f2bf(val);
            }
        }
    }
}

// ---------------------------------------------------------------------------
// Attention K/V chunk staging: K via async-to-LDS, V transposed via DS stores.
// ---------------------------------------------------------------------------
__device__ __forceinline__ void stage_kv(
    const u16* __restrict__ Kb, const u16* __restrict__ Vb,
    int b, int h, int ch, u16* KsBuf, u16* VsBuf, int tid)
{
    if (tid < 128) {                        // async-stage K chunk
        int key = tid >> 2, seg = (tid & 3) * 16;
        const u16* src = Kb + (size_t)(b * 2048 + ch * 32 + key) * 1024 + h * 64 + seg;
        unsigned lo = lds_off(&KsBuf[key * 72 + seg]);
        async_b128(lo,      src);
        async_b128(lo + 16, src + 8);
    } else {                                // stage V chunk (transposed)
        int t2 = tid - 128;
        int key = t2 >> 2, seg = (t2 & 3) * 16;
        const u16* src = Vb + (size_t)(b * 2048 + ch * 32 + key) * 1024 + h * 64 + seg;
#pragma unroll
        for (int j = 0; j < 16; ++j) VsBuf[(seg + j) * 40 + key] = src[j];
    }
}

// ---------------------------------------------------------------------------
// Flash attention: grid (S/128, H, B), 8 waves; wave owns 16 q-rows.
// Double-buffered chunk staging; softmax reductions in DPP (VALU only).
// ---------------------------------------------------------------------------
__global__ __launch_bounds__(256, 1) void attn_fwd(
    const u16* __restrict__ Q, const u16* __restrict__ Kb,
    const u16* __restrict__ Vb, const int* __restrict__ mask,
    u16* __restrict__ Ctx)
{
    __shared__ u16 Ks[2][32 * 72];     // [key][d], pitch 72 u16 = 144B
    __shared__ u16 Vs[2][64 * 40];     // transposed [d][key], pitch 40 u16 = 80B
    __shared__ u16 Ps[8 * 16 * 32];    // per-wave P staging, pitch 32 u16 = 64B

    const int tid  = threadIdx.x;
    const int lane = tid & 31;
    const int w    = tid >> 5;
    const int l    = lane & 15;
    const int hi   = lane >> 4;
    const int b    = blockIdx.z;
    const int h    = blockIdx.y;
    const int q0   = blockIdx.x * 128 + w * 16;

    const u16* qrow = Q + (size_t)(b * 2048 + q0 + l) * 1024 + h * 64;
    FragU aq0, aq1;
    aq0.q[0] = *(const uint4v*)(qrow + hi * 8);
    aq0.q[1] = *(const uint4v*)(qrow + 16 + hi * 8);
    aq1.q[0] = *(const uint4v*)(qrow + 32 + hi * 8);
    aq1.q[1] = *(const uint4v*)(qrow + 48 + hi * 8);

    float mrow[8], lrow[8];
#pragma unroll
    for (int r = 0; r < 8; ++r) { mrow[r] = -3.0e38f; lrow[r] = 0.f; }
    v8f o0 = {}, o1 = {}, o2 = {}, o3 = {};
    u16* Pw = &Ps[w * 512];

    // prologue: stage chunk 0 into buffer 0
    stage_kv(Kb, Vb, b, h, 0, Ks[0], Vs[0], tid);
    wait_async0();
    __syncthreads();

    int cur = 0;
    for (int ch = 0; ch < 64; ++ch) {          // 64 chunks of 32 keys
        // prefetch next chunk into the other buffer (overlaps compute)
        if (ch + 1 < 64)
            stage_kv(Kb, Vb, b, h, ch + 1, Ks[cur ^ 1], Vs[cur ^ 1], tid);

        const u16* Kc = Ks[cur];
        const u16* Vc = Vs[cur];

        // ---- scores: S = Q(16x64) @ K^T(64x32) via 4 WMMAs
        v8f s0 = {}, s1 = {};
        {
            FragU bk;
            bk.q[0] = *(const uint4v*)&Kc[l * 72 + hi * 8];
            bk.q[1] = *(const uint4v*)&Kc[l * 72 + 16 + hi * 8];
            s0 = wmma_bf16(aq0, bk, s0);
            bk.q[0] = *(const uint4v*)&Kc[l * 72 + 32 + hi * 8];
            bk.q[1] = *(const uint4v*)&Kc[l * 72 + 48 + hi * 8];
            s0 = wmma_bf16(aq1, bk, s0);
            int k1 = 16 + l;
            bk.q[0] = *(const uint4v*)&Kc[k1 * 72 + hi * 8];
            bk.q[1] = *(const uint4v*)&Kc[k1 * 72 + 16 + hi * 8];
            s1 = wmma_bf16(aq0, bk, s1);
            bk.q[0] = *(const uint4v*)&Kc[k1 * 72 + 32 + hi * 8];
            bk.q[1] = *(const uint4v*)&Kc[k1 * 72 + 48 + hi * 8];
            s1 = wmma_bf16(aq1, bk, s1);
        }

        // ---- scale + mask + online softmax update (DPP reductions)
        const int m0v = mask[b * 2048 + ch * 32 + l];
        const int m1v = mask[b * 2048 + ch * 32 + 16 + l];
        float p0[8], p1[8];
#pragma unroll
        for (int r = 0; r < 8; ++r) {
            float a = s0[r] * 0.125f; if (m0v == 0) a = -1e9f;
            float e = s1[r] * 0.125f; if (m1v == 0) e = -1e9f;
            float mx = red16_max(fmaxf(a, e));
            float mnew  = fmaxf(mrow[r], mx);
            float scale = __expf(mrow[r] - mnew);
            mrow[r] = mnew;
            float e0 = __expf(a - mnew), e1 = __expf(e - mnew);
            float rsum = red16_sum(e0 + e1);
            lrow[r] = lrow[r] * scale + rsum;
            p0[r] = e0; p1[r] = e1;
            o0[r] *= scale; o1[r] *= scale; o2[r] *= scale; o3[r] *= scale;
        }

        // ---- transpose P (D-layout -> A-layout) through per-wave LDS
#pragma unroll
        for (int r = 0; r < 8; ++r) {
            Pw[(r + hi * 8) * 32 + l]      = f2bf(p0[r]);
            Pw[(r + hi * 8) * 32 + 16 + l] = f2bf(p1[r]);
        }
        asm volatile("s_wait_dscnt 0" ::: "memory");
        FragU pA;
        pA.q[0] = *(const uint4v*)&Pw[l * 32 + hi * 8];
        pA.q[1] = *(const uint4v*)&Pw[l * 32 + 16 + hi * 8];

        // ---- O += P(16x32) @ V(32x64) via 4 WMMAs
        {
            FragU bv;
            bv.q[0] = *(const uint4v*)&Vc[l * 40 + hi * 8];
            bv.q[1] = *(const uint4v*)&Vc[l * 40 + 16 + hi * 8];
            o0 = wmma_bf16(pA, bv, o0);
            bv.q[0] = *(const uint4v*)&Vc[(16 + l) * 40 + hi * 8];
            bv.q[1] = *(const uint4v*)&Vc[(16 + l) * 40 + 16 + hi * 8];
            o1 = wmma_bf16(pA, bv, o1);
            bv.q[0] = *(const uint4v*)&Vc[(32 + l) * 40 + hi * 8];
            bv.q[1] = *(const uint4v*)&Vc[(32 + l) * 40 + 16 + hi * 8];
            o2 = wmma_bf16(pA, bv, o2);
            bv.q[0] = *(const uint4v*)&Vc[(48 + l) * 40 + hi * 8];
            bv.q[1] = *(const uint4v*)&Vc[(48 + l) * 40 + 16 + hi * 8];
            o3 = wmma_bf16(pA, bv, o3);
        }

        wait_async0();
        __syncthreads();
        cur ^= 1;
    }

    // ---- finalize: O / l, write bf16 ctx in [token][1024] layout
#pragma unroll
    for (int r = 0; r < 8; ++r) {
        float inv = 1.0f / lrow[r];
        size_t rowoff = (size_t)(b * 2048 + q0 + r + hi * 8) * 1024 + h * 64;
        Ctx[rowoff + l]      = f2bf(o0[r] * inv);
        Ctx[rowoff + 16 + l] = f2bf(o1[r] * inv);
        Ctx[rowoff + 32 + l] = f2bf(o2[r] * inv);
        Ctx[rowoff + 48 + l] = f2bf(o3[r] * inv);
    }
}

// ---------------------------------------------------------------------------
// launch
// ---------------------------------------------------------------------------
extern "C" void kernel_launch(void* const* d_in, const int* in_sizes, int n_in,
                              void* d_out, int out_size, void* d_ws, size_t ws_size,
                              hipStream_t stream) {
    (void)in_sizes; (void)n_in; (void)out_size; (void)ws_size;
    const float* x    = (const float*)d_in[0];
    const int*   mask = (const int*)d_in[1];
    const float* wq = (const float*)d_in[2];  const float* bq = (const float*)d_in[3];
    const float* wk = (const float*)d_in[4];  const float* bk = (const float*)d_in[5];
    const float* wv = (const float*)d_in[6];  const float* bv = (const float*)d_in[7];
    const float* wo = (const float*)d_in[8];  const float* bo = (const float*)d_in[9];
    const float* w1 = (const float*)d_in[10]; const float* b1 = (const float*)d_in[11];
    const float* w2 = (const float*)d_in[12]; const float* b2 = (const float*)d_in[13];
    const float* g1 = (const float*)d_in[14]; const float* be1 = (const float*)d_in[15];
    const float* g2 = (const float*)d_in[16]; const float* be2 = (const float*)d_in[17];

    char* ws = (char*)d_ws;
    const size_t MB = 1024 * 1024;
    u16*   wqb  = (u16*)(ws + 0 * MB);    // transposed [N][K] bf16
    u16*   wkb  = (u16*)(ws + 2 * MB);
    u16*   wvb  = (u16*)(ws + 4 * MB);
    u16*   wob  = (u16*)(ws + 6 * MB);
    u16*   w1b  = (u16*)(ws + 8 * MB);    // [4096][1024]
    u16*   w2b  = (u16*)(ws + 16 * MB);   // [1024][4096]
    u16*   xnb  = (u16*)(ws + 24 * MB);   // bf16 LN output (reused for both LNs)
    u16*   qb   = (u16*)(ws + 32 * MB);
    u16*   kb   = (u16*)(ws + 40 * MB);
    u16*   vb   = (u16*)(ws + 48 * MB);
    u16*   ctxb = (u16*)(ws + 56 * MB);
    float* x1   = (float*)(ws + 64 * MB); // fp32 residual after MHA
    u16*   hb   = (u16*)(ws + 80 * MB);   // bf16 FFN hidden

    // weights -> bf16, transposed to [N][K]
    cvt_transpose_bf16<<<dim3(32, 32),  256, 0, stream>>>(wq, wqb, 1024, 1024);
    cvt_transpose_bf16<<<dim3(32, 32),  256, 0, stream>>>(wk, wkb, 1024, 1024);
    cvt_transpose_bf16<<<dim3(32, 32),  256, 0, stream>>>(wv, wvb, 1024, 1024);
    cvt_transpose_bf16<<<dim3(32, 32),  256, 0, stream>>>(wo, wob, 1024, 1024);
    cvt_transpose_bf16<<<dim3(128, 32), 256, 0, stream>>>(w1, w1b, 1024, 4096);
    cvt_transpose_bf16<<<dim3(32, 128), 256, 0, stream>>>(w2, w2b, 4096, 1024);

    // LN1
    layernorm_bf16<<<4096, 256, 0, stream>>>(x, g1, be1, xnb);

    // Q/K/V projections (bf16 out)
    gemm_bf16_wmma<0, 0, 0><<<dim3(8, 32), 256, 0, stream>>>(
        xnb, wqb, bq, nullptr, nullptr, qb, 4096, 1024, 1024);
    gemm_bf16_wmma<0, 0, 0><<<dim3(8, 32), 256, 0, stream>>>(
        xnb, wkb, bk, nullptr, nullptr, kb, 4096, 1024, 1024);
    gemm_bf16_wmma<0, 0, 0><<<dim3(8, 32), 256, 0, stream>>>(
        xnb, wvb, bv, nullptr, nullptr, vb, 4096, 1024, 1024);

    // attention
    attn_fwd<<<dim3(16, 16, 2), 256, 0, stream>>>(qb, kb, vb, mask, ctxb);

    // O projection + residual (fp32 out)
    gemm_bf16_wmma<0, 1, 1><<<dim3(8, 32), 256, 0, stream>>>(
        ctxb, wob, bo, x, x1, nullptr, 4096, 1024, 1024);

    // LN2
    layernorm_bf16<<<4096, 256, 0, stream>>>(x1, g2, be2, xnb);

    // FFN1 (ReLU, bf16 out)
    gemm_bf16_wmma<1, 0, 0><<<dim3(32, 32), 256, 0, stream>>>(
        xnb, w1b, b1, nullptr, nullptr, hb, 4096, 4096, 1024);

    // FFN2 + residual -> d_out (fp32)
    gemm_bf16_wmma<0, 1, 1><<<dim3(8, 32), 256, 0, stream>>>(
        hb, w2b, b2, x1, (float*)d_out, nullptr, 4096, 1024, 4096);
}
